// CoNe_82995948028056
// MI455X (gfx1250) — matrix operand order repaired and checked by reading, not compiled, not run on hardware
//
#include <hip/hip_runtime.h>

// ---------------------------------------------------------------------------
// Feature probes (toolchain-dependent CDNA5 builtins)
// ---------------------------------------------------------------------------
#if defined(__has_builtin)
#  if __has_builtin(__builtin_amdgcn_global_load_async_to_lds_b128)
#    define HAS_ASYNC_LDS 1
#  endif
#  if __has_builtin(__builtin_amdgcn_s_wait_asynccnt)
#    define HAS_WAIT_ASYNC 1
#  endif
#  if __has_builtin(__builtin_amdgcn_tensor_load_to_lds) && \
      __has_builtin(__builtin_amdgcn_s_wait_tensorcnt) && (__clang_major__ >= 22)
#    define HAS_TDM 1
#  endif
#endif
#ifndef HAS_ASYNC_LDS
#define HAS_ASYNC_LDS 0
#endif
#ifndef HAS_WAIT_ASYNC
#define HAS_WAIT_ASYNC 0
#endif
#ifndef HAS_TDM
#define HAS_TDM 0
#endif

// ---------------------------------------------------------------------------
// Types for CDNA5 WMMA (wave32)
// ---------------------------------------------------------------------------
typedef __attribute__((ext_vector_type(16))) _Float16     v16h;
typedef __attribute__((ext_vector_type(8)))  _Float16     v8h;
typedef __attribute__((ext_vector_type(8)))  float        v8f;
typedef __attribute__((ext_vector_type(4)))  unsigned int v4u;
typedef __attribute__((ext_vector_type(8)))  int          v8i;
typedef __attribute__((ext_vector_type(4)))  int          v4i;

// Address-space-qualified b128 element types for the async Global->LDS builtin
typedef __attribute__((address_space(1))) v4i g_v4i;   // global
typedef __attribute__((address_space(3))) v4i l_v4i;   // LDS

#define WMMA_F32_16x16x32_F16(a, b, c) \
    __builtin_amdgcn_wmma_f32_16x16x32_f16(false, (a), false, (b), (short)0, (c), false, false)

__device__ inline void async_fence() {
#if HAS_ASYNC_LDS
#  if HAS_WAIT_ASYNC
    __builtin_amdgcn_s_wait_asynccnt(0);
#  else
    asm volatile("s_wait_asynccnt 0" ::: "memory");
#  endif
#endif
}

#if HAS_TDM
// Issue one TDM 2-D tile load (rows x row_len f16 elements, row stride in elements)
// D# packing per CDNA5 ISA section 8.3/8.4.
__device__ inline void tdm_load_2d_to_lds(const void* gsrc, unsigned lds_addr,
                                          unsigned row_len_elts, unsigned rows,
                                          unsigned stride_elts) {
    unsigned long long ga = (unsigned long long)gsrc;
    v4u g0;
    g0[0] = 1u;                                                 // count=1, user mode
    g0[1] = lds_addr;                                           // lds_addr
    g0[2] = (unsigned)(ga & 0xFFFFFFFFull);                     // global_addr lo
    g0[3] = (unsigned)((ga >> 32) & 0x01FFFFFFull) | (2u << 30); // global_addr hi | type=2
    v8i g1;
    g1[0] = (int)(1u << 16);                                    // data_size = 2 bytes
    g1[1] = (int)((row_len_elts & 0xFFFFu) << 16);              // tensor_dim0[15:0]
    g1[2] = (int)(((row_len_elts >> 16) & 0xFFFFu) |            // tensor_dim0[31:16]
                  ((rows & 0xFFFFu) << 16));                    // tensor_dim1[15:0]
    g1[3] = (int)(((rows >> 16) & 0xFFFFu) |                    // tensor_dim1[31:16]
                  ((row_len_elts & 0xFFFFu) << 16));            // tile_dim0 = row_len
    g1[4] = (int)(rows & 0xFFFFu);                              // tile_dim1 = rows, tile_dim2 = 0
    g1[5] = (int)stride_elts;                                   // tensor_dim0_stride[31:0]
    g1[6] = 0;                                                  // stride hi
    g1[7] = 0;
    v4i z4 = {0, 0, 0, 0};
#  if __clang_major__ >= 23
    v8i z8 = {0, 0, 0, 0, 0, 0, 0, 0};
    __builtin_amdgcn_tensor_load_to_lds(g0, g1, z4, z4, z8, 0);
#  else
    __builtin_amdgcn_tensor_load_to_lds(g0, g1, z4, z4, 0);
#  endif
}
#endif

// ---------------------------------------------------------------------------
// Fragment loaders per ISA 7.12.2 (wave32)
// ---------------------------------------------------------------------------
__device__ inline v16h load_a_frag(const _Float16* Atile, int lda) {
    int lane = threadIdx.x & 31;
    int m    = lane & 15;
    int hi   = lane >> 4;
    const _Float16* row = Atile + (size_t)m * lda + hi * 8;
    v8h lo = *(const v8h*)(row);
    v8h hh = *(const v8h*)(row + 16);
    v16h a;
#pragma unroll
    for (int j = 0; j < 8; ++j) { a[j] = lo[j]; a[8 + j] = hh[j]; }
    return a;
}

// B (32x16 f16, K x N row-major): lane%16 = N, elements = K in table order (8h+s, 16+8h+s)
__device__ inline v16h load_b_frag(const _Float16* Btile, int ldb) {
    int lane = threadIdx.x & 31;
    int n    = lane & 15;
    int hi   = lane >> 4;
    v16h b;
#pragma unroll
    for (int j = 0; j < 8; ++j) {
        b[j]     = Btile[(size_t)(hi * 8 + j) * ldb + n];
        b[8 + j] = Btile[(size_t)(16 + hi * 8 + j) * ldb + n];
    }
    return b;
}

// ---------------------------------------------------------------------------
// Generic GEMM: C = act(scale*(A@B + bias [+ residual])) ; A,B f16; C f32/f16
// Block: 256 threads = 8 waves; tile 128x64. B tiles double-buffered in LDS,
// filled with async global->LDS copies (ASYNCcnt) when available.
// ---------------------------------------------------------------------------
__global__ __launch_bounds__(256)
void gemm_f16_kernel(const _Float16* __restrict__ A, const _Float16* __restrict__ B,
                     const float* __restrict__ bias, const float* __restrict__ residual,
                     float* __restrict__ Cf32, _Float16* __restrict__ Cf16,
                     int M, int N, int K, int ldc, float out_scale, int relu)
{
    __shared__ _Float16 Bs[2][32 * 64];
    int wave = threadIdx.x >> 5;
    int lane = threadIdx.x & 31;
    int m0 = blockIdx.y * 128 + wave * 16;
    int n0 = blockIdx.x * 64;

    int tid  = threadIdx.x;
    int crow = tid >> 3;            // 0..31 : K-row within tile
    int ccol = (tid & 7) * 8;       // 0..56 : 8-half (16B) group

    auto stage = [&](int kk, int buf) {
        const _Float16* gsrc = B + (size_t)(kk + crow) * N + n0 + ccol;
        _Float16* ldst = &Bs[buf][crow * 64 + ccol];
#if HAS_ASYNC_LDS
        __builtin_amdgcn_global_load_async_to_lds_b128(
            (g_v4i*)const_cast<_Float16*>(gsrc), (l_v4i*)ldst, 0, 0);
#else
        *(v8h*)ldst = *(const v8h*)gsrc;
#endif
    };

    v8f acc[4] = {};
    const _Float16* Arow = A + (size_t)m0 * K;

    stage(0, 0);
    async_fence();
    __syncthreads();

    for (int kk = 0; kk < K; kk += 32) {
        int buf = (kk >> 5) & 1;
        if (kk + 32 < K) {
            stage(kk + 32, buf ^ 1);
            __builtin_prefetch(Arow + kk + 32, 0, 1);   // global_prefetch_b8
        }
        v16h a = load_a_frag(Arow + kk, K);
#pragma unroll
        for (int t = 0; t < 4; ++t) {
            v16h b = load_b_frag(&Bs[buf][16 * t], 64);
            acc[t] = WMMA_F32_16x16x32_F16(a, b, acc[t]);
        }
        if (kk + 32 < K) async_fence();
        __syncthreads();
    }

    int cn = lane & 15, hi = lane >> 4;
#pragma unroll
    for (int t = 0; t < 4; ++t) {
        int col = n0 + 16 * t + cn;
        float bv = bias ? bias[col] : 0.f;
#pragma unroll
        for (int r = 0; r < 8; ++r) {
            int row = m0 + r + 8 * hi;
            float v = acc[t][r] + bv;
            if (residual) v += residual[(size_t)row * ldc + col];
            v *= out_scale;
            if (relu && v < 0.f) v = 0.f;
            if (Cf32) Cf32[(size_t)row * ldc + col] = v;
            if (Cf16) Cf16[(size_t)row * ldc + col] = (_Float16)v;
        }
    }
}

// ---------------------------------------------------------------------------
// Flash attention pass 1: streaming row max / sum(exp) per 16-query block.
// q pre-scaled by 1/sqrt(C). kT is C x N (transposed keys, f16).
// ---------------------------------------------------------------------------
__global__ __launch_bounds__(32)
void attn_stats_kernel(const _Float16* __restrict__ q, const _Float16* __restrict__ kT,
                       float* __restrict__ rowmax, float* __restrict__ rowsum, int Nrows)
{
    int m0 = blockIdx.x * 16;
    int lane = threadIdx.x & 31;
    v16h qa[8];
#pragma unroll
    for (int f = 0; f < 8; ++f) qa[f] = load_a_frag(q + (size_t)m0 * 256 + f * 32, 256);

    float rm[8], rs[8];
#pragma unroll
    for (int r = 0; r < 8; ++r) { rm[r] = -3.0e38f; rs[r] = 0.f; }

    for (int n0 = 0; n0 < Nrows; n0 += 16) {
        if (n0 + 16 < Nrows)
            __builtin_prefetch(kT + (size_t)lane * Nrows + n0 + 16, 0, 1);
        v8f s = {};
#pragma unroll
        for (int f = 0; f < 8; ++f) {
            v16h b = load_b_frag(kT + (size_t)(f * 32) * Nrows + n0, Nrows);
            s = WMMA_F32_16x16x32_F16(qa[f], b, s);
        }
#pragma unroll
        for (int r = 0; r < 8; ++r) {
            float sv = s[r];
            float nm = fmaxf(rm[r], sv);
            rs[r] = rs[r] * __expf(rm[r] - nm) + __expf(sv - nm);
            rm[r] = nm;
        }
    }
#pragma unroll
    for (int mask = 1; mask < 16; mask <<= 1) {
#pragma unroll
        for (int r = 0; r < 8; ++r) {
            float om = __shfl_xor(rm[r], mask, 32);
            float os = __shfl_xor(rs[r], mask, 32);
            float nm = fmaxf(rm[r], om);
            rs[r] = rs[r] * __expf(rm[r] - nm) + os * __expf(om - nm);
            rm[r] = nm;
        }
    }
    int hi = lane >> 4;
    if ((lane & 15) == 0) {
#pragma unroll
        for (int r = 0; r < 8; ++r) {
            rowmax[m0 + r + 8 * hi] = rm[r];
            rowsum[m0 + r + 8 * hi] = rs[r];
        }
    }
}

// ---------------------------------------------------------------------------
// Flash attention pass 2: O = softmax(q kT) @ V.
// 4 waves / 16 queries. Wave w computes S for key slab w; P relayed through
// LDS; V slab (64x256) staged by the Tensor Data Mover when available.
// ---------------------------------------------------------------------------
__global__ __launch_bounds__(128)
void attn_out_kernel(const _Float16* __restrict__ q, const _Float16* __restrict__ kT,
                     const _Float16* __restrict__ v, const float* __restrict__ rowmax,
                     const float* __restrict__ rowsum, float* __restrict__ O, int Nrows)
{
    __shared__ _Float16 P[16 * 64];
#if HAS_TDM
    __shared__ _Float16 Vs[64 * 256];
#endif
    int wave = threadIdx.x >> 5;
    int lane = threadIdx.x & 31;
    int m0 = blockIdx.x * 16;
    int cn = lane & 15, hi = lane >> 4;

    v16h qa[8];
#pragma unroll
    for (int f = 0; f < 8; ++f) qa[f] = load_a_frag(q + (size_t)m0 * 256 + f * 32, 256);

    float rm[8], ri[8];
#pragma unroll
    for (int r = 0; r < 8; ++r) {
        int row = m0 + r + 8 * hi;
        rm[r] = rowmax[row];
        ri[r] = 1.f / fmaxf(rowsum[row], 1e-12f);
    }

    v8f acc[4] = {};
    for (int kb = 0; kb < Nrows; kb += 64) {
        int key0 = kb + wave * 16;
        v8f s = {};
#pragma unroll
        for (int f = 0; f < 8; ++f) {
            v16h b = load_b_frag(kT + (size_t)(f * 32) * Nrows + key0, Nrows);
            s = WMMA_F32_16x16x32_F16(qa[f], b, s);
        }
        __syncthreads();                 // prior iteration's LDS reads complete
#if HAS_TDM
        if (wave == 0) {                 // TDM stages the 64x256 V slab into LDS
            unsigned lds = (unsigned)(unsigned long long)
                           (__attribute__((address_space(3))) _Float16*)Vs;
            tdm_load_2d_to_lds(v + (size_t)kb * 256, lds, 256u, 64u, 256u);
            __builtin_amdgcn_s_wait_tensorcnt(0);
        }
#endif
#pragma unroll
        for (int r = 0; r < 8; ++r)
            P[(r + 8 * hi) * 64 + wave * 16 + cn] = (_Float16)(__expf(s[r] - rm[r]) * ri[r]);
        __syncthreads();                 // P (and Vs) visible to all waves
#pragma unroll
        for (int ks = 0; ks < 2; ++ks) {
            v16h a = load_a_frag(P + ks * 32, 64);
#pragma unroll
            for (int t = 0; t < 4; ++t) {
#if HAS_TDM
                v16h b = load_b_frag(Vs + (size_t)(ks * 32) * 256 + wave * 64 + 16 * t, 256);
#else
                v16h b = load_b_frag(v + (size_t)(kb + ks * 32) * 256 + wave * 64 + 16 * t, 256);
#endif
                acc[t] = WMMA_F32_16x16x32_F16(a, b, acc[t]);
            }
        }
    }
#pragma unroll
    for (int t = 0; t < 4; ++t)
#pragma unroll
        for (int r = 0; r < 8; ++r)
            O[(size_t)(m0 + r + 8 * hi) * 256 + wave * 64 + 16 * t + cn] = acc[t][r];
}

// ---------------------------------------------------------------------------
// Elementwise / reduction helper kernels
// ---------------------------------------------------------------------------
__global__ void fill_kernel(float* p, float v, int n) {
    int i = blockIdx.x * blockDim.x + threadIdx.x;
    if (i < n) p[i] = v;
}

__global__ void convert_f16_kernel(const float* __restrict__ src, _Float16* __restrict__ dst, int n) {
    int i = blockIdx.x * blockDim.x + threadIdx.x;
    if (i < n) dst[i] = (_Float16)src[i];
}

__global__ void bn_partial_kernel(const float* __restrict__ X, int M, int C, int slices,
                                  float* __restrict__ partial) {
    int c = threadIdx.x;
    int s = blockIdx.x;
    int rows = (M + slices - 1) / slices;
    int r0 = s * rows, r1 = r0 + rows;
    if (r1 > M) r1 = M;
    float su = 0.f, sq = 0.f;
    for (int r = r0; r < r1; ++r) {
        float v = X[(size_t)r * C + c];
        su += v; sq += v * v;
    }
    partial[(size_t)(2 * s) * C + c]     = su;
    partial[(size_t)(2 * s + 1) * C + c] = sq;
}

__global__ void bn_finalize_kernel(const float* __restrict__ partial, int slices, int M, int C,
                                   float* __restrict__ mean, float* __restrict__ var) {
    int c = threadIdx.x;
    float su = 0.f, sq = 0.f;
    for (int i = 0; i < slices; ++i) {
        su += partial[(size_t)(2 * i) * C + c];
        sq += partial[(size_t)(2 * i + 1) * C + c];
    }
    float mu = su / (float)M;
    mean[c] = mu;
    var[c]  = sq / (float)M - mu * mu;
}

__global__ void bn_apply_kernel(const float* __restrict__ X, const float* __restrict__ mean,
                                const float* __restrict__ var, const float* __restrict__ g,
                                const float* __restrict__ b, float* __restrict__ Yf32,
                                _Float16* __restrict__ Yf16, int total, int C, int relu) {
    int i = blockIdx.x * blockDim.x + threadIdx.x;
    if (i >= total) return;
    int c = i % C;
    float v = (X[i] - mean[c]) * rsqrtf(var[c] + 1e-5f) * g[c] + b[c];
    if (relu && v < 0.f) v = 0.f;
    if (Yf32) Yf32[i] = v;
    if (Yf16) Yf16[i] = (_Float16)v;
}

__global__ void deg_kernel(const int* __restrict__ dst, const float* __restrict__ w,
                           float* __restrict__ deg, int E) {
    int e = blockIdx.x * blockDim.x + threadIdx.x;
    if (e < E) atomicAdd(&deg[dst[e]], w[e]);
}

__global__ void dinv_kernel(const float* __restrict__ deg, float* __restrict__ dinv, int n) {
    int i = blockIdx.x * blockDim.x + threadIdx.x;
    if (i < n) dinv[i] = rsqrtf(deg[i]);
}

__global__ __launch_bounds__(256)
void agg_kernel(const int* __restrict__ src, const int* __restrict__ dst,
                const float* __restrict__ w, const float* __restrict__ dinv,
                const float* __restrict__ hw, float* __restrict__ agg) {
    int e = blockIdx.x;
    int c = threadIdx.x;
    int s = src[e], d = dst[e];
    float coef = dinv[s] * w[e] * dinv[d];
    atomicAdd(&agg[(size_t)d * 256 + c], hw[(size_t)s * 256 + c] * coef);
}

__global__ void local_kernel(const float* __restrict__ agg, const float* __restrict__ hw,
                             const float* __restrict__ dinv, const float* __restrict__ b_gcn,
                             const float* __restrict__ h, float* __restrict__ out, int total) {
    int i = blockIdx.x * blockDim.x + threadIdx.x;
    if (i >= total) return;
    int row = i >> 8;           // C = 256
    int c = i & 255;
    float di = dinv[row];
    out[i] = agg[i] + hw[i] * di * di + b_gcn[c] + h[i];
}

__global__ void add_kernel(const float* __restrict__ a, const float* __restrict__ b,
                           float* __restrict__ y, _Float16* __restrict__ y16, int n) {
    int i = blockIdx.x * blockDim.x + threadIdx.x;
    if (i >= n) return;
    float v = a[i] + b[i];
    y[i] = v;
    y16[i] = (_Float16)v;
}

__global__ void transpose_f16_kernel(const _Float16* __restrict__ in, _Float16* __restrict__ out,
                                     int R, int C) {
    int i = blockIdx.x * blockDim.x + threadIdx.x;
    if (i >= R * C) return;
    int r = i / C, c = i % C;
    out[(size_t)c * R + r] = in[i];
}

// Structure branch: one block per query edge; 32 sorted neighbors; per-channel masked softmax.
__global__ __launch_bounds__(256)
void structure_kernel(const int* __restrict__ eli, const int* __restrict__ nbr,
                      const float* __restrict__ z, const float* __restrict__ t_ptr,
                      float* __restrict__ out, int Q) {
    __shared__ int tn[32];
    __shared__ int msk[32];
    int qi = blockIdx.x;
    int tid = threadIdx.x;
    int a = eli[qi];        // edge_label_index[0]
    int b = eli[Q + qi];    // edge_label_index[1]
    if (tid < 32) {
        int tv = nbr[(size_t)a * 32 + tid];
        tn[tid] = tv;
        const int* srow = nbr + (size_t)b * 32;
        int lo = 0, hiB = 32;                 // searchsorted (left)
        while (lo < hiB) { int mid = (lo + hiB) >> 1; if (srow[mid] < tv) lo = mid + 1; else hiB = mid; }
        int pos = lo < 31 ? lo : 31;
        msk[tid] = (srow[pos] == tv) ? 1 : 0;
    }
    __syncthreads();
    float t = *t_ptr;
    int c = tid;
    float zv[32];
#pragma unroll
    for (int d = 0; d < 32; ++d) zv[d] = z[(size_t)tn[d] * 256 + c];
    float m = -1e30f;
#pragma unroll
    for (int d = 0; d < 32; ++d) if (msk[d]) m = fmaxf(m, t * zv[d]);
    float s = 0.f, acc = 0.f;
#pragma unroll
    for (int d = 0; d < 32; ++d)
        if (msk[d]) { float e = __expf(t * zv[d] - m); s += e; acc += e * zv[d]; }
    out[(size_t)qi * 256 + c] = acc / fmaxf(s, 1e-12f);
}

__global__ void content_kernel(const int* __restrict__ eli, const float* __restrict__ z,
                               _Float16* __restrict__ out, int Q) {
    int i = blockIdx.x * blockDim.x + threadIdx.x;
    if (i >= Q * 256) return;
    int qi = i >> 8, c = i & 255;
    int a = eli[qi], b = eli[Q + qi];
    out[i] = (_Float16)(z[(size_t)a * 256 + c] * z[(size_t)b * 256 + c]);
}

__global__ __launch_bounds__(256)
void final_dot_kernel(const float* __restrict__ hmid, const float* __restrict__ w,
                      const float* __restrict__ b, float* __restrict__ out) {
    __shared__ float red[256];
    int qi = blockIdx.x;
    int c = threadIdx.x;
    red[c] = hmid[(size_t)qi * 256 + c] * w[c];
    __syncthreads();
    for (int s = 128; s > 0; s >>= 1) {
        if (c < s) red[c] += red[c + s];
        __syncthreads();
    }
    if (c == 0) out[qi] = red[0] + b[0];
}

// ---------------------------------------------------------------------------
// Host orchestration
// ---------------------------------------------------------------------------
extern "C" void kernel_launch(void* const* d_in, const int* in_sizes, int n_in,
                              void* d_out, int out_size, void* d_ws, size_t ws_size,
                              hipStream_t stream) {
    constexpr int N_ = 8192, D_ = 32, Q_ = 4096, CIN = 128, C_ = 256, C2 = 512;
    constexpr int E_ = N_ * D_;

    const float* x          = (const float*)d_in[0];
    const int*   edge_index = (const int*)d_in[1];
    const float* edge_w     = (const float*)d_in[2];
    const int*   eli        = (const int*)d_in[3];
    const int*   nbr        = (const int*)d_in[4];
    const float* pe_gamma   = (const float*)d_in[5];
    const float* pe_beta    = (const float*)d_in[6];
    const float* W_pe       = (const float*)d_in[7];
    const float* b_pe       = (const float*)d_in[8];
    const float* W_gcn      = (const float*)d_in[9];
    const float* b_gcn      = (const float*)d_in[10];
    const float* bn1_g      = (const float*)d_in[11];
    const float* bn1_b      = (const float*)d_in[12];
    const float* Wq         = (const float*)d_in[13];
    const float* bq         = (const float*)d_in[14];
    const float* Wk         = (const float*)d_in[15];
    const float* bk         = (const float*)d_in[16];
    const float* Wv         = (const float*)d_in[17];
    const float* bv         = (const float*)d_in[18];
    const float* Wo         = (const float*)d_in[19];
    const float* bo         = (const float*)d_in[20];
    const float* bn2_g      = (const float*)d_in[21];
    const float* bn2_b      = (const float*)d_in[22];
    const float* W_m1       = (const float*)d_in[23];
    const float* b_m1       = (const float*)d_in[24];
    const float* W_m2       = (const float*)d_in[25];
    const float* b_m2       = (const float*)d_in[26];
    const float* bn3_g      = (const float*)d_in[27];
    const float* bn3_b      = (const float*)d_in[28];
    const float* W_cm       = (const float*)d_in[29];
    const float* b_cm       = (const float*)d_in[30];
    const float* W_sm       = (const float*)d_in[31];
    const float* b_sm       = (const float*)d_in[32];
    const float* W_o1       = (const float*)d_in[33];
    const float* b_o1       = (const float*)d_in[34];
    const float* bno_g      = (const float*)d_in[35];
    const float* bno_b      = (const float*)d_in[36];
    const float* W_o2       = (const float*)d_in[37];
    const float* b_o2       = (const float*)d_in[38];
    const float* t_ptr      = (const float*)d_in[39];

    const int* e_src = edge_index;        // row 0
    const int* e_dst = edge_index + E_;   // row 1

    // ---- workspace allocator ----
    char* base = (char*)d_ws;
    size_t off = 0;
    auto alloc = [&](size_t bytes) -> void* {
        off = (off + 255) & ~(size_t)255;
        void* p = base + off;
        off += bytes;
        return p;
    };

    _Float16* wpe16  = (_Float16*)alloc((size_t)CIN * C_ * 2);
    _Float16* wgcn16 = (_Float16*)alloc((size_t)C_ * C_ * 2);
    _Float16* wq16   = (_Float16*)alloc((size_t)C_ * C_ * 2);
    _Float16* wk16   = (_Float16*)alloc((size_t)C_ * C_ * 2);
    _Float16* wv16   = (_Float16*)alloc((size_t)C_ * C_ * 2);
    _Float16* wo16   = (_Float16*)alloc((size_t)C_ * C_ * 2);
    _Float16* wm116  = (_Float16*)alloc((size_t)C_ * C2 * 2);
    _Float16* wm216  = (_Float16*)alloc((size_t)C2 * C_ * 2);
    _Float16* wcm16  = (_Float16*)alloc((size_t)C_ * C_ * 2);
    _Float16* wsm16  = (_Float16*)alloc((size_t)C_ * C_ * 2);
    _Float16* wo116  = (_Float16*)alloc((size_t)C2 * C_ * 2);

    float* part  = (float*)alloc((size_t)64 * 2 * C_ * 4);
    float* meanv = (float*)alloc((size_t)C_ * 4);
    float* varv  = (float*)alloc((size_t)C_ * 4);

    _Float16* x16   = (_Float16*)alloc((size_t)N_ * CIN * 2);
    float*    h     = (float*)alloc((size_t)N_ * C_ * 4);
    _Float16* h16   = (_Float16*)alloc((size_t)N_ * C_ * 2);
    float*    deg   = (float*)alloc((size_t)N_ * 4);
    float*    dinv  = (float*)alloc((size_t)N_ * 4);
    float*    hw    = (float*)alloc((size_t)N_ * C_ * 4);
    float*    agg   = (float*)alloc((size_t)N_ * C_ * 4);
    float*    pre   = (float*)alloc((size_t)N_ * C_ * 4);   // reused for all BN preludes
    float*    h1    = (float*)alloc((size_t)N_ * C_ * 4);
    _Float16* q16   = (_Float16*)alloc((size_t)N_ * C_ * 2);
    _Float16* k16   = (_Float16*)alloc((size_t)N_ * C_ * 2);
    _Float16* kT16  = (_Float16*)alloc((size_t)N_ * C_ * 2);
    _Float16* v16b  = (_Float16*)alloc((size_t)N_ * C_ * 2);
    float*    rowmx = (float*)alloc((size_t)N_ * 4);
    float*    rowsm = (float*)alloc((size_t)N_ * 4);
    float*    Obuf  = (float*)alloc((size_t)N_ * C_ * 4);
    _Float16* O16   = (_Float16*)alloc((size_t)N_ * C_ * 2);
    float*    h2    = (float*)alloc((size_t)N_ * C_ * 4);
    float*    z0    = (float*)alloc((size_t)N_ * C_ * 4);
    _Float16* z016  = (_Float16*)alloc((size_t)N_ * C_ * 2);
    _Float16* m116  = (_Float16*)alloc((size_t)N_ * C2 * 2);
    float*    z     = (float*)alloc((size_t)N_ * C_ * 4);
    float*    strct = (float*)alloc((size_t)Q_ * C_ * 4);
    _Float16* str16 = (_Float16*)alloc((size_t)Q_ * C_ * 2);
    _Float16* con16 = (_Float16*)alloc((size_t)Q_ * C_ * 2);
    _Float16* hcat16= (_Float16*)alloc((size_t)Q_ * C2 * 2);
    float*    hmid  = (float*)alloc((size_t)Q_ * C_ * 4);

    float* out = (float*)d_out;

    auto conv = [&](const float* src, _Float16* dst, int n) {
        convert_f16_kernel<<<(n + 255) / 256, 256, 0, stream>>>(src, dst, n);
    };
    auto gemm = [&](const _Float16* A, const _Float16* B, const float* bias, const float* resid,
                    float* Cf, _Float16* Ch, int M, int Ncol, int K, int ldc,
                    float scale, int relu) {
        dim3 g(Ncol / 64, M / 128);
        gemm_f16_kernel<<<g, 256, 0, stream>>>(A, B, bias, resid, Cf, Ch, M, Ncol, K, ldc, scale, relu);
    };
    auto bn = [&](const float* X, int M, int Ccols, const float* g, const float* b,
                  float* Yf, _Float16* Yh, int relu) {
        bn_partial_kernel<<<64, Ccols, 0, stream>>>(X, M, Ccols, 64, part);
        bn_finalize_kernel<<<1, Ccols, 0, stream>>>(part, 64, M, Ccols, meanv, varv);
        int total = M * Ccols;
        bn_apply_kernel<<<(total + 255) / 256, 256, 0, stream>>>(X, meanv, varv, g, b, Yf, Yh,
                                                                 total, Ccols, relu);
    };

    // ---- weights -> f16 ----
    conv(W_pe, wpe16, CIN * C_);
    conv(W_gcn, wgcn16, C_ * C_);
    conv(Wq, wq16, C_ * C_);
    conv(Wk, wk16, C_ * C_);
    conv(Wv, wv16, C_ * C_);
    conv(Wo, wo16, C_ * C_);
    conv(W_m1, wm116, C_ * C2);
    conv(W_m2, wm216, C2 * C_);
    conv(W_cm, wcm16, C_ * C_);
    conv(W_sm, wsm16, C_ * C_);
    conv(W_o1, wo116, C2 * C_);

    // ---- h = BN(x) @ W_pe + b_pe ----
    bn(x, N_, CIN, pe_gamma, pe_beta, nullptr, x16, 0);
    gemm(x16, wpe16, b_pe, nullptr, h, h16, N_, C_, CIN, C_, 1.f, 0);

    // ---- GCN branch ----
    fill_kernel<<<(N_ + 255) / 256, 256, 0, stream>>>(deg, 1.0f, N_);
    deg_kernel<<<(E_ + 255) / 256, 256, 0, stream>>>(e_dst, edge_w, deg, E_);
    dinv_kernel<<<(N_ + 255) / 256, 256, 0, stream>>>(deg, dinv, N_);
    gemm(h16, wgcn16, nullptr, nullptr, hw, nullptr, N_, C_, C_, C_, 1.f, 0);
    fill_kernel<<<(N_ * C_ + 255) / 256, 256, 0, stream>>>(agg, 0.0f, N_ * C_);
    agg_kernel<<<E_, 256, 0, stream>>>(e_src, e_dst, edge_w, dinv, hw, agg);
    local_kernel<<<(N_ * C_ + 255) / 256, 256, 0, stream>>>(agg, hw, dinv, b_gcn, h, pre, N_ * C_);
    bn(pre, N_, C_, bn1_g, bn1_b, h1, nullptr, 0);

    // ---- attention branch ----
    gemm(h16, wq16, bq, nullptr, nullptr, q16, N_, C_, C_, C_, 0.0625f /*1/sqrt(256)*/, 0);
    gemm(h16, wk16, bk, nullptr, nullptr, k16, N_, C_, C_, C_, 1.f, 0);
    gemm(h16, wv16, bv, nullptr, nullptr, v16b, N_, C_, C_, C_, 1.f, 0);
    transpose_f16_kernel<<<(N_ * C_ + 255) / 256, 256, 0, stream>>>(k16, kT16, N_, C_);
    attn_stats_kernel<<<N_ / 16, 32, 0, stream>>>(q16, kT16, rowmx, rowsm, N_);
    attn_out_kernel<<<N_ / 16, 128, 0, stream>>>(q16, kT16, v16b, rowmx, rowsm, Obuf, N_);
    conv(Obuf, O16, N_ * C_);
    gemm(O16, wo16, bo, h, pre, nullptr, N_, C_, C_, C_, 1.f, 0);
    bn(pre, N_, C_, bn2_g, bn2_b, h2, nullptr, 0);

    // ---- MLP + residual ----
    add_kernel<<<(N_ * C_ + 255) / 256, 256, 0, stream>>>(h1, h2, z0, z016, N_ * C_);
    gemm(z016, wm116, b_m1, nullptr, nullptr, m116, N_, C2, C_, C2, 1.f, 1);
    gemm(m116, wm216, b_m2, z0, pre, nullptr, N_, C_, C2, C_, 1.f, 0);
    bn(pre, N_, C_, bn3_g, bn3_b, z, nullptr, 0);

    // ---- query heads ----
    structure_kernel<<<Q_, 256, 0, stream>>>(eli, nbr, z, t_ptr, strct, Q_);
    conv(strct, str16, Q_ * C_);
    content_kernel<<<(Q_ * C_ + 255) / 256, 256, 0, stream>>>(eli, z, con16, Q_);
    gemm(con16, wcm16, b_cm, nullptr, nullptr, hcat16,       Q_, C_, C_, C2, 1.f, 0);
    gemm(str16, wsm16, b_sm, nullptr, nullptr, hcat16 + C_,  Q_, C_, C_, C2, 1.f, 0);
    gemm(hcat16, wo116, b_o1, nullptr, pre, nullptr, Q_, C_, C2, C_, 1.f, 0);
    bn(pre, Q_, C_, bno_g, bno_b, hmid, nullptr, 1);
    final_dot_kernel<<<Q_, 256, 0, stream>>>(hmid, W_o2, b_o2, out);
}
